// QuantLinear_72799695667838
// MI455X (gfx1250) — compile-verified, matrix operand order checked by reading, not run on hardware
//
#include <hip/hip_runtime.h>

// ---------- CDNA5 WMMA types ----------
typedef __attribute__((ext_vector_type(16))) _Float16 v16h;
typedef __attribute__((ext_vector_type(8)))  float    v8f;

// Tile config: block = 256 threads = 8 waves (wave32).
// Block tile 128(M) x 128(N), K staged 64 at a time, double-buffered LDS.
// Wave grid 4(M) x 2(N): each wave computes 32x64 = 2x4 WMMA 16x16 tiles.
// One 16x16x32 WMMA spans exactly one E8M0 scale group (32 K), so the group
// scale is applied exactly, in f32, on the WMMA partial product.
#define BM 128
#define BN 128
#define BK 64
#define LDS_KD 36          // row stride in dwords = (64 + 8 pad) halves / 2 -> conflict-free b128 reads

static __device__ __forceinline__ unsigned int pack2h(float a, float b) {
    _Float16 ha = (_Float16)a, hb = (_Float16)b;
    unsigned short ua = __builtin_bit_cast(unsigned short, ha);
    unsigned short ub = __builtin_bit_cast(unsigned short, hb);
    return (unsigned int)ua | ((unsigned int)ub << 16);
}

// One packed byte (two int4 sign-magnitude nibbles) -> two UNSCALED f16 mags.
// True value = ±mag * 0.25 * 2^(s-127); the 0.25 and 2^(s-127) are folded into
// the per-group f32 scale applied after the WMMA.
static __device__ __forceinline__ unsigned int dq2(int b) {
    short lo = (short)(b & 7);
    if (b & 0x8)  lo = (short)-lo;
    short hi = (short)((b >> 4) & 7);
    if (b & 0x80) hi = (short)-hi;
    _Float16 flo = (_Float16)lo;
    _Float16 fhi = (_Float16)hi;
    return (unsigned int)__builtin_bit_cast(unsigned short, flo) |
           ((unsigned int)__builtin_bit_cast(unsigned short, fhi) << 16);
}

// Load a 16x32 f16 fragment (8 VGPRs) from LDS: two b128 reads per lane,
// per the ISA 16-bit A/B layout (VGPR0-3 = K base, VGPR4-7 = K base + 16).
static __device__ __forceinline__ v16h ld_frag(const unsigned int* base) {
    struct P { uint4 a, b; } p;
    p.a = *(const uint4*)(base);
    p.b = *(const uint4*)(base + 8);   // +16 halves
    return __builtin_bit_cast(v16h, p);
}

__global__ void __launch_bounds__(256)
mxfp4_linear_wmma(const float* __restrict__ x,
                  const int*   __restrict__ wp,   // [O, I/2] one packed byte per int32
                  const int*   __restrict__ ws,   // [O, I/32] E8M0 scales
                  const float* __restrict__ bias, // [O]
                  float*       __restrict__ out,  // [T, O]
                  int T, int I, int O) {
    __shared__ __align__(16) unsigned int sA[2 * BM * LDS_KD];  // f16 pairs
    __shared__ __align__(16) unsigned int sB[2 * BN * LDS_KD];
    __shared__ float sS[2 * 2 * BN];                            // f32 group scales

    const int tid  = threadIdx.x;
    const int lane = tid & 31;
    const int wid  = tid >> 5;
    const int wave_m = wid & 3;   // 0..3 -> 32 rows each
    const int wave_n = wid >> 2;  // 0..1 -> 64 cols each

    const int blockn = blockIdx.x;   // N tile
    const int blockm = blockIdx.y;   // M tile

    // ---- loader thread mapping ----
    const int a_r = tid >> 4;        // 0..15 : row within a 16-row pass
    const int a_c = tid & 15;        // float4 column within 64-K tile
    const int b_n    = tid >> 1;     // 0..127 : weight row n
    const int b_half = tid & 1;      // which 32-K half (= one scale group)

    const size_t Ihalf  = (size_t)I >> 1;
    const size_t Igroup = (size_t)I >> 5;
    const int n_glob = blockn * BN + b_n;
    const int* wp_row = wp + (size_t)n_glob * Ihalf;
    const int* ws_row = ws + (size_t)n_glob * Igroup;

    // ---- per-lane fragment LDS offsets (dwords) ----
    const int lrow = lane & 15;
    const int lkhi = (lane >> 4) << 2;   // +8 halves = +4 dwords for upper half-wave
    int aoff[2], boff[4];
#pragma unroll
    for (int mt = 0; mt < 2; ++mt)
        aoff[mt] = (wave_m * 32 + mt * 16 + lrow) * LDS_KD + lkhi;
#pragma unroll
    for (int nt = 0; nt < 4; ++nt)
        boff[nt] = (wave_n * 64 + nt * 16 + lrow) * LDS_KD + lkhi;

    v8f acc[2][4];
#pragma unroll
    for (int mt = 0; mt < 2; ++mt)
#pragma unroll
        for (int nt = 0; nt < 4; ++nt)
            acc[mt][nt] = v8f{};

    // ---- register staging for the double-buffered pipeline ----
    float4 aReg[8];
    int4   wReg[4];
    int    sReg;

    auto issue_loads = [&](int k0) {
#pragma unroll
        for (int it = 0; it < 8; ++it) {
            const int m = it * 16 + a_r;
            aReg[it] = *(const float4*)(x + (size_t)(blockm * BM + m) * I + k0 + a_c * 4);
        }
        const int4* pw = (const int4*)(wp_row + (k0 >> 1) + b_half * 16);
#pragma unroll
        for (int q = 0; q < 4; ++q) wReg[q] = pw[q];
        sReg = ws_row[(k0 >> 5) + b_half];
        if (k0 + BK < I) {   // prefetch one tile further ahead (global_prefetch_b8)
            __builtin_prefetch(x + (size_t)(blockm * BM + a_r) * I + k0 + BK + a_c * 4, 0, 1);
            __builtin_prefetch(wp_row + ((k0 + BK) >> 1) + b_half * 16, 0, 1);
        }
    };

    auto commit_to_lds = [&](int buf) {
        unsigned int* dA = sA + buf * (BM * LDS_KD);
        unsigned int* dB = sB + buf * (BN * LDS_KD);
        // exact f32 group scale: 2^(s-127) * 0.25
        sS[buf * (2 * BN) + b_half * BN + b_n] = exp2f((float)(sReg - 129));
#pragma unroll
        for (int it = 0; it < 8; ++it) {
            const int m = it * 16 + a_r;
            dA[m * LDS_KD + a_c * 2]     = pack2h(aReg[it].x, aReg[it].y);
            dA[m * LDS_KD + a_c * 2 + 1] = pack2h(aReg[it].z, aReg[it].w);
        }
        unsigned int* db = dB + b_n * LDS_KD + b_half * 16;
#pragma unroll
        for (int q = 0; q < 4; ++q) {
            db[q * 4 + 0] = dq2(wReg[q].x);
            db[q * 4 + 1] = dq2(wReg[q].y);
            db[q * 4 + 2] = dq2(wReg[q].z);
            db[q * 4 + 3] = dq2(wReg[q].w);
        }
    };

    auto compute = [&](int buf) {
        const unsigned int* pA = sA + buf * (BM * LDS_KD);
        const unsigned int* pB = sB + buf * (BN * LDS_KD);
        const float*        pS = sS + buf * (2 * BN);
#pragma unroll
        for (int ks = 0; ks < 2; ++ks) {
            const int kd = ks * 16;   // 32 halves = 16 dwords
            v16h af[2], bf[4];
            float sc[4];
#pragma unroll
            for (int mt = 0; mt < 2; ++mt)
                af[mt] = ld_frag(pA + aoff[mt] + kd);
#pragma unroll
            for (int nt = 0; nt < 4; ++nt) {
                bf[nt] = ld_frag(pB + boff[nt] + kd);
                sc[nt] = pS[ks * BN + wave_n * 64 + nt * 16 + lrow];
            }
            const v8f z = {};
            v8f d[2][4];
#pragma unroll
            for (int mt = 0; mt < 2; ++mt)
#pragma unroll
                for (int nt = 0; nt < 4; ++nt)
                    d[mt][nt] = __builtin_amdgcn_wmma_f32_16x16x32_f16(
                        false, af[mt], false, bf[nt], (short)0, z, false, false);
            // per-group E8M0 scale applied exactly in f32; co-executes with WMMA
#pragma unroll
            for (int mt = 0; mt < 2; ++mt)
#pragma unroll
                for (int nt = 0; nt < 4; ++nt)
                    acc[mt][nt] += d[mt][nt] * sc[nt];
        }
    };

    // ---- pipelined main loop: one barrier per K-tile ----
    issue_loads(0);
    commit_to_lds(0);
    int cur = 0;
    for (int k0 = 0; k0 < I; k0 += BK) {
        __syncthreads();                       // LDS buf[cur] ready, buf[cur^1] consumed
        const bool has_next = (k0 + BK < I);
        if (has_next) issue_loads(k0 + BK);    // global latency overlaps WMMAs below
        compute(cur);
        if (has_next) commit_to_lds(cur ^ 1);
        cur ^= 1;
    }

    // ---- epilogue: bias + store (lanes 0-15 row r, lanes 16-31 row r+8) ----
    const int orow_base = blockm * BM + wave_m * 32 + ((lane >> 4) << 3);
    const int ocol_base = blockn * BN + wave_n * 64 + (lane & 15);
#pragma unroll
    for (int nt = 0; nt < 4; ++nt) {
        const int col = ocol_base + nt * 16;
        const float bv = bias[col];
#pragma unroll
        for (int mt = 0; mt < 2; ++mt) {
#pragma unroll
            for (int r = 0; r < 8; ++r) {
                const int row = orow_base + mt * 16 + r;
                out[(size_t)row * O + col] = acc[mt][nt][r] + bv;
            }
        }
    }
}

extern "C" void kernel_launch(void* const* d_in, const int* in_sizes, int n_in,
                              void* d_out, int out_size, void* d_ws, size_t ws_size,
                              hipStream_t stream) {
    const float* x    = (const float*)d_in[0];
    const int*   wp   = (const int*)d_in[1];
    const int*   ws   = (const int*)d_in[2];
    const float* bias = (const float*)d_in[3];
    float*       out  = (float*)d_out;

    const long O = in_sizes[3];
    const long I = (2L * in_sizes[1]) / O;   // wp is [O, I/2]
    const long T = (long)in_sizes[0] / I;    // x  is [T, I]

    dim3 grid((unsigned)(O / BN), (unsigned)(T / BM));
    mxfp4_linear_wmma<<<grid, 256, 0, stream>>>(x, wp, ws, bias, out,
                                                (int)T, (int)I, (int)O);
}
